// SegmentRecurrentHead_40956808134849
// MI455X (gfx1250) — matrix operand order repaired and checked by reading, not compiled, not run on hardware
//
#include <hip/hip_runtime.h>
#include <hip/hip_bf16.h>

typedef __attribute__((ext_vector_type(16))) _Float16 v16h;
typedef __attribute__((ext_vector_type(8)))  float    v8f;
typedef _Float16 half_t;

// ---------------------------------------------------------------------------
// CDNA5 async global->LDS copy (16B per lane), tracked by ASYNCcnt.
// Generic pointers to __shared__ keep the LDS byte offset in their low 32 bits.
// ---------------------------------------------------------------------------
__device__ __forceinline__ unsigned lds_addr32(const void* p) {
    return (unsigned)(uintptr_t)p;
}
__device__ __forceinline__ void async_copy_b128(unsigned lds_off, const half_t* gptr) {
    unsigned long long ga = (unsigned long long)(uintptr_t)gptr;
    asm volatile("global_load_async_to_lds_b128 %0, %1, off"
                 :: "v"(lds_off), "v"(ga)
                 : "memory");
}
__device__ __forceinline__ void wait_async_le8() {
    asm volatile("s_wait_asynccnt 0x8" ::: "memory");
}
__device__ __forceinline__ void wait_async_0() {
    asm volatile("s_wait_asynccnt 0x0" ::: "memory");
}

// ---------------------------------------------------------------------------
// fp32 -> fp16 convert (grid-stride)
// ---------------------------------------------------------------------------
__global__ void cvt_f32_f16_kernel(const float* __restrict__ src,
                                   half_t* __restrict__ dst, long long n) {
    long long i = (long long)blockIdx.x * blockDim.x + threadIdx.x;
    long long stride = (long long)gridDim.x * blockDim.x;
    for (; i < n; i += stride) dst[i] = (half_t)src[i];
}

// h_tile[b] = concat(h_prev[b], h[b]) -> f16, (8,2048,1024)
__global__ void concat_cvt_kernel(const float* __restrict__ h_prev,
                                  const float* __restrict__ h,
                                  half_t* __restrict__ out, long long n) {
    long long i = (long long)blockIdx.x * blockDim.x + threadIdx.x;
    long long stride = (long long)gridDim.x * blockDim.x;
    for (; i < n; i += stride) {
        long long d = i & 1023;
        long long r = (i >> 10) & 2047;
        long long b = i >> 21;
        float v;
        if (r < 1024) v = h_prev[(b << 20) + (r << 10) + d];
        else          v = h     [(b << 20) + ((r - 1024) << 10) + d];
        out[i] = (half_t)v;
    }
}

// ---------------------------------------------------------------------------
// Tiled WMMA GEMM:  C(MxN) = A(MxK,f16) @ W(NxK,f16)^T + bias
// Workgroup tile 128x128, 8 waves (4 row x 2 col), wave tile 32x64 (2x4 WMMA).
// K-chunk 64 (2 wmma K-steps per stage), LDS double-buffered, filled with
// global_load_async_to_lds_b128 (ASYNCcnt), rows padded to 72 halfs (banks).
// lda == K, ldw == K for all operands.
// ---------------------------------------------------------------------------
#define KC 64
#define SP 72   // padded LDS row stride in halfs

__global__ __launch_bounds__(256)
void gemm_wmma_kernel(const half_t* __restrict__ A, long long strideAb,
                      const half_t* __restrict__ W, long long strideWb,
                      const float*  __restrict__ bias,
                      float*  __restrict__ Cf, long long strideCfb, int ldcf,
                      half_t* __restrict__ Ch, long long strideChb, int ldch,
                      half_t* __restrict__ Ct, long long strideCtb, int ldct,
                      int M, int N, int K)
{
    __shared__ __align__(32) half_t As[2][128 * SP];
    __shared__ __align__(32) half_t Ws[2][128 * SP];

    const int bz = blockIdx.z;
    A += (long long)bz * strideAb;
    W += (long long)bz * strideWb;

    const int m0   = blockIdx.x * 128;
    const int n0   = blockIdx.y * 128;
    const int tid  = threadIdx.x;
    const int wave = tid >> 5;
    const int lane = tid & 31;
    const int wr   = wave & 3;   // wave row 0..3 (32 rows each)
    const int wc   = wave >> 2;  // wave col 0..1 (64 cols each)
    const int lr   = lane & 15;
    const int lg   = lane >> 4;  // lane group 0/1

    // staging geometry: 128 rows x 8 uint4 per row = 1024 slots, 4 per thread
    const int srow = tid >> 3;       // base row stepped by 32
    const int sc8  = tid & 7;        // uint4 column within row

    // ---- zero-fill OOB rows of both buffers once (A and W tails) ----
    const uint4 z4 = make_uint4(0u, 0u, 0u, 0u);
    #pragma unroll
    for (int i = 0; i < 4; ++i) {
        int row = srow + i * 32;
        if (m0 + row >= M) {
            *(uint4*)(&As[0][row * SP + sc8 * 8]) = z4;
            *(uint4*)(&As[1][row * SP + sc8 * 8]) = z4;
        }
        if (n0 + row >= N) {
            *(uint4*)(&Ws[0][row * SP + sc8 * 8]) = z4;
            *(uint4*)(&Ws[1][row * SP + sc8 * 8]) = z4;
        }
    }

    const unsigned ldsA0 = lds_addr32(&As[0][0]);
    const unsigned ldsA1 = lds_addr32(&As[1][0]);
    const unsigned ldsW0 = lds_addr32(&Ws[0][0]);
    const unsigned ldsW1 = lds_addr32(&Ws[1][0]);

    // issue one stage of async copies into buffer `buf` for k-offset k0
    auto issue_stage = [&](int k0, int buf) {
        const unsigned la = (buf ? ldsA1 : ldsA0);
        const unsigned lw = (buf ? ldsW1 : ldsW0);
        #pragma unroll
        for (int i = 0; i < 4; ++i) {
            int row = srow + i * 32;
            int gm = m0 + row;
            if (gm < M)
                async_copy_b128(la + (unsigned)(row * SP + sc8 * 8) * 2,
                                A + (long long)gm * K + k0 + sc8 * 8);
        }
        #pragma unroll
        for (int i = 0; i < 4; ++i) {
            int row = srow + i * 32;
            int gn = n0 + row;
            if (gn < N)
                async_copy_b128(lw + (unsigned)(row * SP + sc8 * 8) * 2,
                                W + (long long)gn * K + k0 + sc8 * 8);
        }
    };

    v8f acc[2][4] = {};
    union F { uint4 q[2]; v16h v; };

    const int nstage = K / KC;
    issue_stage(0, 0);

    for (int t = 0; t < nstage; ++t) {
        const int cur = t & 1;
        if (t + 1 < nstage) {
            issue_stage((t + 1) * KC, cur ^ 1);
            wait_async_le8();   // 8 in flight for next stage; current stage done
        } else {
            wait_async_0();
        }
        __syncthreads();        // current stage visible to all waves

        const half_t* Ab = &As[cur][0];
        const half_t* Wb = &Ws[cur][0];
        #pragma unroll
        for (int ks = 0; ks < 2; ++ks) {
            const int kb = ks * 32;
            v16h afrag[2], bfrag[4];
            // A 16x32 frag: lanes0-15 K=[0..7]+[16..23], lanes16-31 K=[8..15]+[24..31]
            #pragma unroll
            for (int mt = 0; mt < 2; ++mt) {
                const half_t* arow = Ab + (wr * 32 + mt * 16 + lr) * SP + kb;
                F f;
                f.q[0] = *(const uint4*)(arow + lg * 8);
                f.q[1] = *(const uint4*)(arow + 16 + lg * 8);
                afrag[mt] = f.v;
            }
            // B 32x16 frag (= W^T): lanes0-15 col=lr K=[0..7]+[16..23]; lanes16-31 K=[8..15]+[24..31]
            #pragma unroll
            for (int nt = 0; nt < 4; ++nt) {
                const half_t* wrow = Wb + (wc * 64 + nt * 16 + lr) * SP + kb;
                F f;
                f.q[0] = *(const uint4*)(wrow + lg * 8);
                f.q[1] = *(const uint4*)(wrow + 16 + lg * 8);
                bfrag[nt] = f.v;
            }
            #pragma unroll
            for (int mt = 0; mt < 2; ++mt)
                #pragma unroll
                for (int nt = 0; nt < 4; ++nt)
                    acc[mt][nt] = __builtin_amdgcn_wmma_f32_16x16x32_f16(
                        false, afrag[mt], false, bfrag[nt],
                        (short)0, acc[mt][nt], false, false);
        }
        __syncthreads();        // all waves done reading `cur`; safe to overwrite
    }

    // ---- epilogue: C layout: VGPR i, lanes0-15 -> (M=i,N=lane); lanes16-31 -> (M=i+8,N=lane-16)
    #pragma unroll
    for (int mt = 0; mt < 2; ++mt) {
        #pragma unroll
        for (int nt = 0; nt < 4; ++nt) {
            int nG = n0 + wc * 64 + nt * 16 + lr;
            if (nG >= N) continue;
            int mBase = m0 + wr * 32 + mt * 16 + (lg ? 8 : 0);
            float bv = bias ? bias[nG] : 0.0f;
            #pragma unroll
            for (int i = 0; i < 8; ++i) {
                int mG = mBase + i;
                if (mG >= M) continue;
                float cv = acc[mt][nt][i] + bv;
                if (Cf) Cf[(long long)bz * strideCfb + (long long)mG * ldcf + nG] = cv;
                if (Ch) Ch[(long long)bz * strideChb + (long long)mG * ldch + nG] = (half_t)cv;
                if (Ct) Ct[(long long)bz * strideCtb + (long long)nG * ldct + mG] = (half_t)cv;
            }
        }
    }
}

// ---------------------------------------------------------------------------
// out[row] = dot(X[row,:], vec)   one wave32 per row
// ---------------------------------------------------------------------------
__global__ __launch_bounds__(256)
void dot_rows_kernel(const half_t* __restrict__ X, const float* __restrict__ vec,
                     float* __restrict__ out, int nrows, int K) {
    int wave = threadIdx.x >> 5;
    int lane = threadIdx.x & 31;
    int row  = blockIdx.x * 8 + wave;
    if (row >= nrows) return;
    const half_t* xr = X + (long long)row * K;
    float acc = 0.0f;
    for (int k = lane; k < K; k += 32) acc += (float)xr[k] * vec[k];
    #pragma unroll
    for (int off = 16; off > 0; off >>= 1) acc += __shfl_xor(acc, off, 32);
    if (lane == 0) out[row] = acc;
}

// ---------------------------------------------------------------------------
// scores assembly + masked softmax per row (b,l); Mk=2048 keys, 256 thr x 8 cols
// ---------------------------------------------------------------------------
__global__ __launch_bounds__(256)
void score_softmax_kernel(const float* __restrict__ S, const half_t* __restrict__ Bfl,
                          const float* __restrict__ third, const float* __restrict__ dall,
                          const float* __restrict__ mask, half_t* __restrict__ P,
                          int L, int Mk, int NB) {
    const int row = blockIdx.x;           // b*L + l
    const int b = row / L, l = row % L;
    const float scale = 0.03125f;         // 1/sqrt(1024)
    const float*  Srow = S     + (long long)row * Mk;
    const half_t* Brow = Bfl   + (long long)row * NB;
    const float*  thr  = third + (long long)b * Mk;
    const float*  mrow = mask  + (long long)b * Mk;

    float sv[8];
    float lmax = -3.0e38f;
    #pragma unroll
    for (int i = 0; i < 8; ++i) {
        int j = threadIdx.x + i * 256;
        int p = (L - 1 - l) + j;
        float s = (Srow[j] + (float)Brow[p] + thr[j] + dall[p]) * scale;
        if (mrow[j] <= 0.0f) s = -1.0e30f;
        sv[i] = s;
        lmax = fmaxf(lmax, s);
    }
    __shared__ float red[256];
    red[threadIdx.x] = lmax; __syncthreads();
    for (int off = 128; off > 0; off >>= 1) {
        if (threadIdx.x < off) red[threadIdx.x] = fmaxf(red[threadIdx.x], red[threadIdx.x + off]);
        __syncthreads();
    }
    float rmax = red[0]; __syncthreads();

    float lsum = 0.0f;
    #pragma unroll
    for (int i = 0; i < 8; ++i) { sv[i] = __expf(sv[i] - rmax); lsum += sv[i]; }
    red[threadIdx.x] = lsum; __syncthreads();
    for (int off = 128; off > 0; off >>= 1) {
        if (threadIdx.x < off) red[threadIdx.x] += red[threadIdx.x + off];
        __syncthreads();
    }
    float inv = 1.0f / red[0];
    half_t* Prow = P + (long long)row * Mk;
    #pragma unroll
    for (int i = 0; i < 8; ++i) {
        int j = threadIdx.x + i * 256;
        Prow[j] = (half_t)(sv[i] * inv);
    }
}

// ---------------------------------------------------------------------------
extern "C" void kernel_launch(void* const* d_in, const int* in_sizes, int n_in,
                              void* d_out, int out_size, void* d_ws, size_t ws_size,
                              hipStream_t stream) {
    const float* h_prev = (const float*)d_in[0];
    const float* h      = (const float*)d_in[1];
    const float* mask   = (const float*)d_in[2];
    const float* Wq_w   = (const float*)d_in[3];
    const float* Wq_b   = (const float*)d_in[4];
    const float* Wke_w  = (const float*)d_in[5];
    const float* Wke_b  = (const float*)d_in[6];
    const float* Wkr_w  = (const float*)d_in[7];
    const float* Wkr_b  = (const float*)d_in[8];
    const float* Wv_w   = (const float*)d_in[9];
    const float* Wv_b   = (const float*)d_in[10];
    const float* u_vec  = (const float*)d_in[11];
    const float* v_vec  = (const float*)d_in[12];
    const float* R      = (const float*)d_in[13];
    float* out = (float*)d_out;

    const long long B = 8, Lq = 1024, Mk = 2048, D = 1024, NP = 3071;

    char* ws = (char*)d_ws;
    size_t off = 0;
    auto alloc = [&](size_t bytes) -> char* {
        char* p = ws + off;
        off = (off + bytes + 255) & ~(size_t)255;
        return p;
    };
    half_t* h16   = (half_t*)alloc(B * Mk * D * 2);
    half_t* wq16  = (half_t*)alloc(D * D * 2);
    half_t* wke16 = (half_t*)alloc(D * D * 2);
    half_t* wkr16 = (half_t*)alloc(D * D * 2);
    half_t* wv16  = (half_t*)alloc(D * D * 2);
    half_t* r16   = (half_t*)alloc(NP * D * 2);
    half_t* q16   = (half_t*)alloc(B * Lq * D * 2);
    half_t* k16   = (half_t*)alloc(B * Mk * D * 2);
    half_t* vT16  = (half_t*)alloc(B * D * Mk * 2);
    half_t* qr16  = (half_t*)alloc(NP * D * 2);
    half_t* b16   = (half_t*)alloc(B * Lq * NP * 2);
    float*  Sqk   = (float*) alloc(B * Lq * Mk * 4);
    half_t* p16   = (half_t*)alloc(B * Lq * Mk * 2);
    float*  dall  = (float*) alloc(NP * 4);
    float*  third = (float*) alloc(B * Mk * 4);
    (void)ws_size; (void)in_sizes; (void)n_in; (void)out_size;

    // ---- stage 1: precision conversion ----
    concat_cvt_kernel<<<4096, 256, 0, stream>>>(h_prev, h, h16, B * Mk * D);
    cvt_f32_f16_kernel<<<2048, 256, 0, stream>>>(Wq_w,  wq16,  D * D);
    cvt_f32_f16_kernel<<<2048, 256, 0, stream>>>(Wke_w, wke16, D * D);
    cvt_f32_f16_kernel<<<2048, 256, 0, stream>>>(Wkr_w, wkr16, D * D);
    cvt_f32_f16_kernel<<<2048, 256, 0, stream>>>(Wv_w,  wv16,  D * D);
    cvt_f32_f16_kernel<<<2048, 256, 0, stream>>>(R,     r16,   NP * D);

    auto gemm = [&](const half_t* Ap, long long sA, const half_t* Wp, long long sW,
                    const float* bp,
                    float* Cf, long long sCf, int ldcf,
                    half_t* Ch, long long sCh, int ldch,
                    half_t* Ct, long long sCt, int ldct,
                    int M, int N, int K, int batch) {
        dim3 grid((M + 127) / 128, (N + 127) / 128, batch);
        gemm_wmma_kernel<<<grid, 256, 0, stream>>>(Ap, sA, Wp, sW, bp,
                                                   Cf, sCf, ldcf, Ch, sCh, ldch,
                                                   Ct, sCt, ldct, M, N, K);
    };

    // ---- stage 2: projections ----
    gemm(h16 + Lq * D, Mk * D, wq16, 0, Wq_b,
         nullptr, 0, 0, q16, Lq * D, (int)D, nullptr, 0, 0,
         (int)Lq, (int)D, (int)D, (int)B);                       // q
    gemm(h16, 0, wke16, 0, Wke_b,
         nullptr, 0, 0, k16, 0, (int)D, nullptr, 0, 0,
         (int)(B * Mk), (int)D, (int)D, 1);                      // k
    gemm(h16, Mk * D, wv16, 0, Wv_b,
         nullptr, 0, 0, nullptr, 0, 0, vT16, D * Mk, (int)Mk,
         (int)Mk, (int)D, (int)D, (int)B);                       // v^T
    gemm(r16, 0, wkr16, 0, Wkr_b,
         nullptr, 0, 0, qr16, 0, (int)D, nullptr, 0, 0,
         (int)NP, (int)D, (int)D, 1);                            // Qr

    // ---- stage 3: B_tile = q @ Qr^T ----
    gemm(q16, 0, qr16, 0, nullptr,
         nullptr, 0, 0, b16, 0, (int)NP, nullptr, 0, 0,
         (int)(B * Lq), (int)NP, (int)D, 1);

    // ---- stage 4: first = q @ k^T (batched, f32) ----
    gemm(q16, Lq * D, k16, Mk * D, nullptr,
         Sqk, Lq * Mk, (int)Mk, nullptr, 0, 0, nullptr, 0, 0,
         (int)Lq, (int)Mk, (int)D, (int)B);

    // ---- stage 5: small dot products ----
    dot_rows_kernel<<<(int)((NP + 7) / 8), 256, 0, stream>>>(qr16, v_vec, dall, (int)NP, (int)D);
    dot_rows_kernel<<<(int)((B * Mk) / 8), 256, 0, stream>>>(k16, u_vec, third, (int)(B * Mk), (int)D);

    // ---- stage 6: assemble scores + softmax -> P ----
    score_softmax_kernel<<<(int)(B * Lq), 256, 0, stream>>>(Sqk, b16, third, dall, mask,
                                                            p16, (int)Lq, (int)Mk, (int)NP);

    // ---- stage 7: out = P @ V (batched, f32 -> d_out) ----
    gemm(p16, Lq * Mk, vT16, D * Mk, nullptr,
         out, Lq * D, (int)D, nullptr, 0, 0, nullptr, 0, 0,
         (int)Lq, (int)D, (int)Mk, (int)B);
}
